// Decoder_25091198943819
// MI455X (gfx1250) — matrix operand order repaired and checked by reading, not compiled, not run on hardware
//
#include <hip/hip_runtime.h>
#include <math.h>

#define H    32
#define KVH  8
#define D    128
#define HID  4096
#define THETA 10000.0f
#define CHUNK 256
#define WAVES 8
#define XSTRIDE 68   // dwords per X-tile row  (16B aligned, bank-staggered)
#define PSTRIDE 68   // dwords per packed-P column (16B aligned, bank-staggered)

typedef __attribute__((ext_vector_type(16))) __bf16       v16bf;
typedef __attribute__((ext_vector_type(8)))  float        v8f;
typedef __attribute__((ext_vector_type(8)))  unsigned int v8u;
typedef __attribute__((ext_vector_type(4)))  unsigned int v4u;

__device__ __forceinline__ unsigned short f2bf(float f) {
    unsigned u = __builtin_bit_cast(unsigned, f);
    u += 0x7FFFu + ((u >> 16) & 1u);          // round-to-nearest-even
    return (unsigned short)(u >> 16);
}
__device__ __forceinline__ unsigned packbf2(float a, float b) {
    return (unsigned)f2bf(a) | ((unsigned)f2bf(b) << 16);
}
__device__ __forceinline__ float bf2f(unsigned short b) {
    unsigned u = ((unsigned)b) << 16;
    return __builtin_bit_cast(float, u);
}
__device__ __forceinline__ float silu_f(float x) { return x / (1.f + __expf(-x)); }

__device__ __forceinline__ v8f wmma_bf16(v16bf a, v16bf b, v8f c) {
    return __builtin_amdgcn_wmma_f32_16x16x32_bf16(false, a, false, b, (short)0, c,
                                                   false, false);
}

// A fragment (16x32 bf16, K-offset k0) from dword-packed row-major X tile.
// lane m=lane&15, hi=lane>>4. VGPR v=0..3: K=k0+hi*8+2v(+1); v=4..7: K=k0+16+hi*8+2(v-4).
// Consecutive K pairs are adjacent ushorts -> two aligned 16B loads, no shifts.
__device__ __forceinline__ v16bf loadA(const unsigned int* Xd, int k0, int lane) {
    int m = lane & 15, hi = lane >> 4;
    const unsigned int* p = Xd + m * XSTRIDE + (k0 >> 1) + hi * 4;
    v4u lo = *(const v4u*)p;         // dwords (k0>>1)+hi*4 .. +3
    v4u hh = *(const v4u*)(p + 8);   // dwords (k0>>1)+8+hi*4 .. +3
    v8u t;
    t[0] = lo[0]; t[1] = lo[1]; t[2] = lo[2]; t[3] = lo[3];
    t[4] = hh[0]; t[5] = hh[1]; t[6] = hh[2]; t[7] = hh[3];
    return __builtin_bit_cast(v16bf, t);
}

// B fragment (32x16 bf16, rows k0..k0+31, cols n0*16..+15) from pair-packed,
// transposed P: Pp[col*PSTRIDE + k/2] = P[k][col] | P[k+1][col]<<16.
// lane n=lane&15, hi=lane>>4. VGPR v=0..3: K=k0+hi*8+2v; v=4..7: K=k0+16+hi*8+2(v-4)
// -> two aligned 16B loads (same K order as loadA).
__device__ __forceinline__ v16bf loadB(const unsigned int* Pp, int k0, int n0, int lane) {
    int n = lane & 15, hi = lane >> 4;
    const unsigned int* p = Pp + (n0 * 16 + n) * PSTRIDE + (k0 >> 1) + hi * 4;
    v4u a = *(const v4u*)p;
    v4u b = *(const v4u*)(p + 8);
    v8u t;
    t[0] = a[0]; t[1] = a[1]; t[2] = a[2]; t[3] = a[3];
    t[4] = b[0]; t[5] = b[1]; t[6] = b[2]; t[7] = b[3];
    return __builtin_bit_cast(v16bf, t);
}

// ---------------------------------------------------------------- GEMV: y = x @ W
__global__ void gemv_kernel(const float* __restrict__ x, const float* __restrict__ W,
                            float* __restrict__ y, int rows, int cols) {
    __shared__ float xs[HID];
    for (int i = threadIdx.x; i < rows; i += blockDim.x) xs[i] = x[i];
    __syncthreads();
    int j = blockIdx.x * blockDim.x + threadIdx.x;
    if (j >= cols) return;
    float acc = 0.f;
    for (int i = 0; i < rows; ++i) acc = fmaf(xs[i], W[(size_t)i * cols + j], acc);
    y[j] = acc;
}

// ---------------------------------------------------------------- q-side: RoPE + hash
__global__ void qside_kernel(const float* __restrict__ q_raw,
                             const float* __restrict__ hp1, const float* __restrict__ hb1,
                             const float* __restrict__ hp2, const float* __restrict__ hb2,
                             float* __restrict__ qr_ws, unsigned* __restrict__ qhash_ws,
                             int S) {
    int h = blockIdx.x, e = threadIdx.x;
    __shared__ float xa[D], ta[D];
    int   fi   = e & 63;
    float invf = __expf(-(float)fi * (__logf(THETA) / 64.f));
    float sn, cs;
    __sincosf((float)S * invf, &sn, &cs);
    float xe = q_raw[h * D + e];
    float xo = (e < 64) ? -q_raw[h * D + e + 64] : q_raw[h * D + e - 64];
    float x  = xe * cs + xo * sn;
    xa[e] = x;
    qr_ws[h * D + e] = x;
    __syncthreads();
    const float* P1 = hp1 + (size_t)h * D * D;
    float acc = 0.f;
    for (int d = 0; d < D; ++d) acc = fmaf(xa[d], P1[d * D + e], acc);
    float t1 = silu_f(acc + hb1[h * D + e]) + x;
    ta[e] = t1;
    __syncthreads();
    const float* P2 = hp2 + (size_t)h * D * D;
    float acc2 = 0.f;
    for (int d = 0; d < D; ++d) acc2 = fmaf(ta[d], P2[d * D + e], acc2);
    float u = acc2 + hb2[h * D + e] + t1;
    unsigned m = __builtin_amdgcn_ballot_w32(u > 0.f);
    if ((threadIdx.x & 31) == 0) qhash_ws[h * 4 + (threadIdx.x >> 5)] = m;
}

// ---------------------------------------------------------------- heavy WMMA kernel
__global__ void __launch_bounds__(256)
hashscore_kernel(const float* __restrict__ key_cache, const float* __restrict__ k_new,
                 const float* __restrict__ hp1, const float* __restrict__ hb1,
                 const float* __restrict__ hp2, const float* __restrict__ hb2,
                 const float* __restrict__ qr_ws, const unsigned* __restrict__ qhash_ws,
                 float* __restrict__ scores_ws, int* __restrict__ drafts_ws,
                 int S, int K, int Kpad) {
    extern __shared__ char smem[];
    unsigned int* P1p = (unsigned int*)smem;                 // 128 * 68 dw = 34816B
    unsigned int* P2p = P1p + D * PSTRIDE;                   // 34816B
    unsigned int* Xall = P2p + D * PSTRIDE;                  // 8 waves * 16*68 dw = 34816B
    unsigned short* khall = (unsigned short*)(Xall + WAVES * 16 * XSTRIDE); // 2048B
    float*    qrs = (float*)(khall + WAVES * 16 * 8);        // 128 f
    float*    b1s = qrs + D;                                 // 128 f
    float*    b2s = b1s + D;                                 // 128 f
    unsigned* qh  = (unsigned*)(b2s + D);                    // 4 u32

    int h    = blockIdx.x;
    int base = blockIdx.y * CHUNK;
    int tid  = threadIdx.x, lane = tid & 31, wave = tid >> 5;

    // Stage P1/P2 transposed + K-pair-packed bf16 (coalesced global reads).
    const float* P1g = hp1 + (size_t)h * D * D;
    const float* P2g = hp2 + (size_t)h * D * D;
    for (int i = tid; i < D * 64; i += blockDim.x) {
        int col = i & 127, kk = i >> 7;                      // kk = k/2
        float a1 = P1g[(2 * kk) * D + col], b1 = P1g[(2 * kk + 1) * D + col];
        float a2 = P2g[(2 * kk) * D + col], b2 = P2g[(2 * kk + 1) * D + col];
        P1p[col * PSTRIDE + kk] = packbf2(a1, b1);
        P2p[col * PSTRIDE + kk] = packbf2(a2, b2);
    }
    if (tid < D) {
        qrs[tid] = qr_ws[h * D + tid];
        b1s[tid] = hb1[h * D + tid];
        b2s[tid] = hb2[h * D + tid];
    }
    if (tid < 4) qh[tid] = qhash_ws[h * 4 + tid];
    __syncthreads();

    unsigned int*   Xd = Xall + wave * 16 * XSTRIDE;
    unsigned short* Xs = (unsigned short*)Xd;                // row stride 2*XSTRIDE ushorts
    unsigned short* kh = khall + wave * 16 * 8;

    int   c   = lane * 4;                       // this lane's 4 columns (0..124)
    float sgn = (c < 64) ? -1.f : 1.f;          // rotate_half sign for these cols
    float invfv[4];
#pragma unroll
    for (int i = 0; i < 4; ++i)
        invfv[i] = __expf(-(float)((c + i) & 63) * (__logf(THETA) / 64.f));

    for (int t = wave; t < 16; t += WAVES) {
        int krow0 = base + t * 16;

        // -------- stage 16 roped rows as bf16, compute raw q.k scores --------
        for (int row = 0; row < 16; ++row) {
            int   kk = krow0 + row;
            float xv[4], yv[4];
            if (kk < S) {
                const float* g = key_cache + ((size_t)h * S + kk) * D;
                float4 a = *(const float4*)(g + c);
                float4 b = *(const float4*)(g + (c ^ 64));
                xv[0] = a.x; xv[1] = a.y; xv[2] = a.z; xv[3] = a.w;
                yv[0] = b.x; yv[1] = b.y; yv[2] = b.z; yv[3] = b.w;
            } else if (kk == S) {
                const float* g = k_new + (size_t)(h >> 2) * D;   // GQA: kv head = h/4
                float4 a = *(const float4*)(g + c);
                float4 b = *(const float4*)(g + (c ^ 64));
                xv[0] = a.x; xv[1] = a.y; xv[2] = a.z; xv[3] = a.w;
                yv[0] = b.x; yv[1] = b.y; yv[2] = b.z; yv[3] = b.w;
            } else {
#pragma unroll
                for (int i = 0; i < 4; ++i) { xv[i] = 0.f; yv[i] = 0.f; }
            }
            float out[4], sp = 0.f;
#pragma unroll
            for (int i = 0; i < 4; ++i) {
                float sn, cs;
                __sincosf((float)kk * invfv[i], &sn, &cs);
                out[i] = xv[i] * cs + sgn * yv[i] * sn;
                sp = fmaf(qrs[c + i], out[i], sp);
            }
            uint2 pk;
            pk.x = packbf2(out[0], out[1]);
            pk.y = packbf2(out[2], out[3]);
            *(uint2*)(Xd + row * XSTRIDE + (c >> 1)) = pk;     // one ds_store_b64
#pragma unroll
            for (int o = 16; o > 0; o >>= 1) sp += __shfl_xor(sp, o, 32);
            if (lane == 0 && kk < K)
                scores_ws[(size_t)h * Kpad + kk] = sp * 0.0883883476483184f; // 1/sqrt(128)
        }

        int n = lane & 15, hi = lane >> 4;
        v8f acc[8];

        // -------- matmul 1: T = silu(X @ P1 + b1) + X   (T overwrites X) --------
        {
            v16bf a0 = loadA(Xd, 0, lane), a1 = loadA(Xd, 32, lane);
            v16bf a2 = loadA(Xd, 64, lane), a3 = loadA(Xd, 96, lane);
#pragma unroll
            for (int n0 = 0; n0 < 8; ++n0) {
                v8f ca = {};
                ca = wmma_bf16(a0, loadB(P1p, 0,  n0, lane), ca);
                ca = wmma_bf16(a1, loadB(P1p, 32, n0, lane), ca);
                ca = wmma_bf16(a2, loadB(P1p, 64, n0, lane), ca);
                ca = wmma_bf16(a3, loadB(P1p, 96, n0, lane), ca);
                acc[n0] = ca;
            }
#pragma unroll
            for (int n0 = 0; n0 < 8; ++n0) {
                int colE = n0 * 16 + n;
#pragma unroll
                for (int r = 0; r < 8; ++r) {
                    int   m  = r + 8 * hi;
                    float x0 = bf2f(Xs[m * (2 * XSTRIDE) + colE]);
                    float tv = silu_f(acc[n0][r] + b1s[colE]) + x0;
                    Xs[m * (2 * XSTRIDE) + colE] = f2bf(tv);
                }
            }
        }

        // -------- matmul 2: U = T @ P2 + b2 + T, khash = sign bits --------
        {
            v16bf a0 = loadA(Xd, 0, lane), a1 = loadA(Xd, 32, lane);
            v16bf a2 = loadA(Xd, 64, lane), a3 = loadA(Xd, 96, lane);
#pragma unroll
            for (int n0 = 0; n0 < 8; ++n0) {
                v8f ca = {};
                ca = wmma_bf16(a0, loadB(P2p, 0,  n0, lane), ca);
                ca = wmma_bf16(a1, loadB(P2p, 32, n0, lane), ca);
                ca = wmma_bf16(a2, loadB(P2p, 64, n0, lane), ca);
                ca = wmma_bf16(a3, loadB(P2p, 96, n0, lane), ca);
                acc[n0] = ca;
            }
#pragma unroll
            for (int n0 = 0; n0 < 8; ++n0) {
                int colE = n0 * 16 + n;
#pragma unroll
                for (int r = 0; r < 8; ++r) {
                    int   m  = r + 8 * hi;
                    float tv = bf2f(Xs[m * (2 * XSTRIDE) + colE]);
                    float u  = acc[n0][r] + b2s[colE] + tv;
                    unsigned msk = __builtin_amdgcn_ballot_w32(u > 0.f);
                    if (lane == 0) {                 // low 16 = row r, high 16 = row r+8
                        kh[r * 8 + n0]       = (unsigned short)(msk & 0xFFFFu);
                        kh[(r + 8) * 8 + n0] = (unsigned short)(msk >> 16);
                    }
                }
            }
        }

        // -------- draft score: 128 - 2*popc(khash ^ qhash) --------
        if (lane < 16) {
            int kk = krow0 + lane;
            if (kk < K) {
                int pop = 0;
#pragma unroll
                for (int j = 0; j < 4; ++j) {
                    unsigned w32 = (unsigned)kh[lane * 8 + 2 * j] |
                                   ((unsigned)kh[lane * 8 + 2 * j + 1] << 16);
                    pop += __popc(w32 ^ qh[j]);
                }
                drafts_ws[(size_t)h * Kpad + kk] = 128 - 2 * pop;
            }
        }
    }
}

// ---------------------------------------------------------------- counting-sort top-k
__global__ void select_kernel(float* __restrict__ scores_ws, const int* __restrict__ drafts_ws,
                              int* __restrict__ keep_ws, float* __restrict__ out_ws,
                              float* __restrict__ denom_ws, int K, int Kpad, int num_remain) {
    int h = blockIdx.x, tid = threadIdx.x, lane = tid & 31, wave = tid >> 5;
    __shared__ unsigned hist[129];
    __shared__ unsigned wcnt[8];
    __shared__ float    wred[8];
    __shared__ int      sBstar, sQuota, sRun;
    __shared__ float    sM;

    const int* dr = drafts_ws + (size_t)h * Kpad;
    float*     sc = scores_ws + (size_t)h * Kpad;
    int*       kp = keep_ws + (size_t)h * Kpad;

    for (int i = tid; i < 129; i += blockDim.x) hist[i] = 0;
    __syncthreads();
    for (int k = tid; k < K; k += blockDim.x)
        atomicAdd(&hist[(128 - dr[k]) >> 1], 1u);      // bin 0 = best (draft=128)
    __syncthreads();
    if (tid == 0) {
        unsigned cum = 0; int B = 128, quota = num_remain;
        for (int b = 0; b <= 128; ++b) {
            unsigned cnt = hist[b];
            if (cum + cnt >= (unsigned)num_remain) { B = b; quota = num_remain - (int)cum; break; }
            cum += cnt;
        }
        sBstar = B; sQuota = quota; sRun = 0;
    }
    __syncthreads();
    int   Bstar = sBstar, quota = sQuota;
    float lmax  = -3.4e38f;

    for (int b0 = 0; b0 < K; b0 += blockDim.x) {
        int k   = b0 + tid;
        int bin = 200;
        if (k < K) bin = (128 - dr[k]) >> 1;
        bool     tie = (bin == Bstar);
        unsigned mb  = __builtin_amdgcn_ballot_w32(tie);
        if (lane == 0) wcnt[wave] = __popc(mb);
        __syncthreads();
        unsigned before = 0, tot = 0;
        for (int w2 = 0; w2 < 8; ++w2) {
            if (w2 < wave) before += wcnt[w2];
            tot += wcnt[w2];
        }
        int rank = sRun + (int)before + __popc(mb & ((1u << lane) - 1u));
        if (k < K) {
            int keep = (bin < Bstar) || (tie && rank < quota);
            kp[k] = keep;
            if (keep) lmax = fmaxf(lmax, sc[k]);
        }
        __syncthreads();
        if (tid == 0) sRun += (int)tot;
        __syncthreads();
    }

#pragma unroll
    for (int o = 16; o > 0; o >>= 1) lmax = fmaxf(lmax, __shfl_xor(lmax, o, 32));
    if (lane == 0) wred[wave] = lmax;
    __syncthreads();
    if (tid == 0) {
        float m = wred[0];
        for (int w2 = 1; w2 < 8; ++w2) m = fmaxf(m, wred[w2]);
        sM = m;
    }
    __syncthreads();
    float m = sM, lsum = 0.f;
    for (int k = tid; k < K; k += blockDim.x) {
        float w = kp[k] ? __expf(sc[k] - m) : 0.f;
        sc[k] = w;                                    // weights overwrite scores
        lsum += w;
    }
#pragma unroll
    for (int o = 16; o > 0; o >>= 1) lsum += __shfl_xor(lsum, o, 32);
    if (lane == 0) wred[wave] = lsum;
    __syncthreads();
    if (tid == 0) {
        float s = 0.f;
        for (int w2 = 0; w2 < 8; ++w2) s += wred[w2];
        denom_ws[h] = s;
    }
    if (tid < D) out_ws[h * D + tid] = 0.f;           // zero accumulator for gather
}

// ---------------------------------------------------------------- sparse weighted V-sum
__global__ void gatherv_kernel(const float* __restrict__ val_cache, const float* __restrict__ v_new,
                               const float* __restrict__ w_ws, float* __restrict__ out_ws,
                               int S, int K, int Kpad, int nranges) {
    int h = blockIdx.x, r = blockIdx.y, d = threadIdx.x;
    int len = (K + nranges - 1) / nranges;
    int k0 = r * len, k1 = min(k0 + len, K);
    const float* w = w_ws + (size_t)h * Kpad;
    float acc = 0.f;
    for (int k = k0; k < k1; ++k) {
        float wk = w[k];
        if (wk != 0.f) {
            float v = (k < S) ? val_cache[((size_t)h * S + k) * D + d]
                              : v_new[(size_t)(h >> 2) * D + d];
            acc = fmaf(wk, v, acc);
        }
    }
    atomicAdd(&out_ws[h * D + d], acc);
}

// ---------------------------------------------------------------- normalize + @Wo
__global__ void outproj_kernel(const float* __restrict__ out_ws, const float* __restrict__ denom_ws,
                               const float* __restrict__ Wo, float* __restrict__ y) {
    __shared__ float vec[HID];
    for (int i = threadIdx.x; i < HID; i += blockDim.x)
        vec[i] = out_ws[i] / denom_ws[i >> 7];
    __syncthreads();
    int j = blockIdx.x * blockDim.x + threadIdx.x;
    float acc = 0.f;
    for (int i = 0; i < HID; ++i) acc = fmaf(vec[i], Wo[(size_t)i * HID + j], acc);
    y[j] = acc;
}

extern "C" void kernel_launch(void* const* d_in, const int* in_sizes, int n_in,
                              void* d_out, int out_size, void* d_ws, size_t ws_size,
                              hipStream_t stream) {
    (void)n_in; (void)out_size; (void)ws_size;
    const float* hs  = (const float*)d_in[0];
    const float* kc  = (const float*)d_in[1];
    const float* vc  = (const float*)d_in[2];
    const float* Wq  = (const float*)d_in[3];
    const float* Wk  = (const float*)d_in[4];
    const float* Wv  = (const float*)d_in[5];
    const float* Wo  = (const float*)d_in[6];
    const float* hp1 = (const float*)d_in[7];
    const float* hb1 = (const float*)d_in[8];
    const float* hp2 = (const float*)d_in[9];
    const float* hb2 = (const float*)d_in[10];
    float* out = (float*)d_out;

    int S      = in_sizes[1] / (H * D);
    int K      = S + 1;
    int chunks = (K + CHUNK - 1) / CHUNK;
    int Kpad   = chunks * CHUNK;
    int mn     = (K < 128) ? K : 128;
    int nr     = K - (int)((double)K * 0.9);
    int num_remain = (nr > mn) ? nr : mn;

    float*    wsf    = (float*)d_ws;
    float*    q_raw  = wsf;                   // HID
    float*    qr     = q_raw + HID;           // H*D
    float*    kn     = qr + H * D;            // KVH*D
    float*    vn     = kn + KVH * D;          // KVH*D
    float*    outv   = vn + KVH * D;          // H*D
    float*    denom  = outv + H * D;          // H
    unsigned* qhash  = (unsigned*)(denom + H);            // H*4
    float*    scores = (float*)(qhash + H * 4);           // H*Kpad (becomes weights)
    int*      drafts = (int*)(scores + (size_t)H * Kpad); // H*Kpad
    int*      keep   = drafts + (size_t)H * Kpad;         // H*Kpad

    gemv_kernel<<<dim3((HID + 255) / 256), dim3(256), 0, stream>>>(hs, Wq, q_raw, HID, HID);
    gemv_kernel<<<dim3((KVH * D + 255) / 256), dim3(256), 0, stream>>>(hs, Wk, kn, HID, KVH * D);
    gemv_kernel<<<dim3((KVH * D + 255) / 256), dim3(256), 0, stream>>>(hs, Wv, vn, HID, KVH * D);

    qside_kernel<<<dim3(H), dim3(D), 0, stream>>>(q_raw, hp1, hb1, hp2, hb2, qr, qhash, S);

    size_t shmem = (size_t)(2 * D * PSTRIDE + WAVES * 16 * XSTRIDE) * sizeof(unsigned)
                 + (size_t)(WAVES * 16 * 8) * sizeof(unsigned short)
                 + (size_t)(3 * D) * sizeof(float)
                 + 4 * sizeof(unsigned);
    hashscore_kernel<<<dim3(H, chunks), dim3(256), shmem, stream>>>(
        kc, kn, hp1, hb1, hp2, hb2, qr, qhash, scores, drafts, S, K, Kpad);

    select_kernel<<<dim3(H), dim3(256), 0, stream>>>(scores, drafts, keep, outv, denom,
                                                     K, Kpad, num_remain);
    gatherv_kernel<<<dim3(H, 8), dim3(D), 0, stream>>>(vc, vn, scores, outv, S, K, Kpad, 8);
    outproj_kernel<<<dim3(HID / 256), dim3(256), 0, stream>>>(outv, denom, Wo, out);
}
